// GraphAttentionEmbedding_45165876085119
// MI455X (gfx1250) — matrix operand order repaired and checked by reading, not compiled, not run on hardware
//
#include <hip/hip_runtime.h>
#include <hip/hip_bf16.h>

typedef __attribute__((ext_vector_type(16))) __bf16 v16bf;
typedef __attribute__((ext_vector_type(2)))  __bf16 v2bf;
typedef __attribute__((ext_vector_type(8)))  float  v8f;
typedef __attribute__((ext_vector_type(4)))  float  v4f;

union Frag { uint4 q[2]; v16bf v; };

__device__ inline unsigned short f32_to_bf16(float f) {
  unsigned u = __float_as_uint(f);
  unsigned r = u + 0x7FFFu + ((u >> 16) & 1u);   // round to nearest even
  return (unsigned short)(r >> 16);
}

__device__ inline unsigned pack_bf16(float a, float b) {
#if __has_builtin(__builtin_amdgcn_cvt_pk_bf16_f32)
  v2bf p = __builtin_amdgcn_cvt_pk_bf16_f32(a, b);
  return __builtin_bit_cast(unsigned, p);
#else
  return (unsigned)f32_to_bf16(a) | ((unsigned)f32_to_bf16(b) << 16);
#endif
}

__device__ inline void atomicAddF(float* p, float v) {
  __hip_atomic_fetch_add(p, v, __ATOMIC_RELAXED, __HIP_MEMORY_SCOPE_AGENT);
}

__device__ inline void atomicMaxF(float* addr, float val) {
  if (val >= 0.0f) atomicMax((int*)addr, __float_as_int(val));
  else             atomicMin((unsigned int*)addr, __float_as_uint(val));
}

// ---- CDNA5 async global->LDS copy (ASYNCcnt) ------------------------------
__device__ __attribute__((always_inline))
inline void async_b128(unsigned lds_off, const void* g) {
  asm volatile("global_load_async_to_lds_b128 %0, %1, off"
               :: "v"(lds_off), "v"((unsigned long long)(size_t)g) : "memory");
}
__device__ __attribute__((always_inline)) inline void wait_async0() {
  asm volatile("s_wait_asynccnt 0x0" ::: "memory");
}
__device__ __attribute__((always_inline)) inline void wait_async1() {
  asm volatile("s_wait_asynccnt 0x1" ::: "memory");
}
__device__ __attribute__((always_inline)) inline void wait_async3() {
  asm volatile("s_wait_asynccnt 0x3" ::: "memory");
}
__device__ __attribute__((always_inline))
inline unsigned lds_off_of(const void* p) {           // low 32 bits = LDS offset
  return (unsigned)(size_t)p;
}

// ---- shared fragment-load + 4x WMMA (layouts per ISA 7.12.2) --------------
__device__ __attribute__((always_inline))
inline void tile_wmma(const unsigned short* bufA, const unsigned short* bufB,
                      int lane, int wave, v8f acc[4]) {
  const int rowL = wave * 16 + (lane & 15);
  const int hi   = lane >> 4;
  Frag a;
  a.q[0] = *(const uint4*)&bufA[rowL * 40 + hi * 8];        // K = hi*8..+7
  a.q[1] = *(const uint4*)&bufA[rowL * 40 + 16 + hi * 8];   // K = 16+hi*8..+7
#pragma unroll
  for (int ct = 0; ct < 4; ++ct) {
    const int col = ct * 16 + (lane & 15);
    Frag b;
    b.q[0] = *(const uint4*)&bufB[col * 40 + hi * 8];
    b.q[1] = *(const uint4*)&bufB[col * 40 + 16 + hi * 8];
    acc[ct] = __builtin_amdgcn_wmma_f32_16x16x32_bf16(
        false, a.v, false, b.v, (short)0, acc[ct], false, false);
  }
}

__device__ __attribute__((always_inline))
inline void epilogue(v8f acc[4], const float* bias, float* C,
                     int mBase, int nBase, int lane, int wave, int M, int Nout) {
  const int hi = lane >> 4;
  const int cL = lane & 15;
#pragma unroll
  for (int ct = 0; ct < 4; ++ct) {
    const int col = nBase + ct * 16 + cL;
    const float bv = bias ? bias[col] : 0.0f;
#pragma unroll
    for (int r = 0; r < 8; ++r) {
      const int row = mBase + wave * 16 + hi * 8 + r;
      if (row < M) C[(size_t)row * Nout + col] = acc[ct][r] + bv;
    }
  }
}

// ---------------------------------------------------------------------------
// Pre-passes: bulk fp32->bf16 convert; weight transpose+convert (one-time)
// ---------------------------------------------------------------------------
__global__ void cvt_f32_to_bf16(const float* __restrict__ in,
                                unsigned short* __restrict__ out, size_t n) {
  size_t i = ((size_t)blockIdx.x * blockDim.x + threadIdx.x) * 8;
  if (i >= n) return;
  const v4f* p = (const v4f*)(in + i);
  v4f a = p[0], b = p[1];
  uint4 o;
  o.x = pack_bf16(a.x, a.y); o.y = pack_bf16(a.z, a.w);
  o.z = pack_bf16(b.x, b.y); o.w = pack_bf16(b.z, b.w);
  *(uint4*)(out + i) = o;
}

__global__ void transpose_cvt(const float* __restrict__ W,
                              unsigned short* __restrict__ Wt, int K, int Nout) {
  int idx = blockIdx.x * blockDim.x + threadIdx.x;     // idx = c*K + k
  if (idx >= K * Nout) return;
  int c = idx / K, k = idx - c * K;
  Wt[idx] = f32_to_bf16(W[(size_t)k * Nout + c]);
}

// ---------------------------------------------------------------------------
// Node GEMM: C = Ab(bf16)[M,K] @ Bt(bf16,[Nout][K])^T + bias.
// Double-buffered LDS; tiles staged entirely with global_load_async_to_lds_b128
// (3 async instr per wave per tile), s_wait_asynccnt pipelining depth 2.
// ---------------------------------------------------------------------------
__global__ __launch_bounds__(256)
void gemm_node(const unsigned short* __restrict__ Ab,
               const unsigned short* __restrict__ Bt,
               const float* __restrict__ bias,
               float* __restrict__ C, int M, int K, int Nout) {
  __shared__ __align__(16) unsigned short lA[2][128 * 40];
  __shared__ __align__(16) unsigned short lB[2][64 * 40];
  const int tid = threadIdx.x, lane = tid & 31, wave = tid >> 5;
  const int mBase = blockIdx.x * 128, nBase = blockIdx.y * 64;
  v8f acc[4] = {};

  const int arow = tid >> 1, aseg = (tid & 1) * 16;
  const int bcol = tid >> 2, bks  = (tid & 3) * 8;
  int gr = mBase + arow; if (gr >= M) gr = M - 1;        // clamp: no OOB DMA

  const unsigned short* gA = Ab + (size_t)gr * K + aseg;
  const unsigned short* gB = Bt + (size_t)(nBase + bcol) * K + bks;

  const unsigned dA[2] = { lds_off_of(&lA[0][arow * 40 + aseg]),
                           lds_off_of(&lA[1][arow * 40 + aseg]) };
  const unsigned dB[2] = { lds_off_of(&lB[0][bcol * 40 + bks]),
                           lds_off_of(&lB[1][bcol * 40 + bks]) };

  const int nk = K / 32;
  // prologue: tiles 0 and 1 in flight
  async_b128(dA[0],      gA);
  async_b128(dA[0] + 16, gA + 8);
  async_b128(dB[0],      gB);
  if (nk > 1) {
    async_b128(dA[1],      gA + 32);
    async_b128(dA[1] + 16, gA + 40);
    async_b128(dB[1],      gB + 32);
  }
  for (int i = 0; i < nk; ++i) {
    if (i + 1 < nk) wait_async3(); else wait_async0();   // current tile resident
    __syncthreads();
    const int buf = i & 1;
    tile_wmma(lA[buf], lB[buf], lane, wave, acc);
    __syncthreads();
    if (i + 2 < nk) {
      const int kc = (i + 2) * 32;
      async_b128(dA[buf],      gA + kc);
      async_b128(dA[buf] + 16, gA + kc + 8);
      async_b128(dB[buf],      gB + kc);
    }
  }
  epilogue(acc, bias, C, mBase, nBase, lane, wave, M, Nout);
}

// ---------------------------------------------------------------------------
// Edge GEMM: C = concat(A0[M,K0], A1[M,K1])(fp32) @ Bt(bf16,[Nout][K])^T.
// B staged with async-to-LDS (double-buffered); A converted fp32->bf16 in
// registers, next tile's global loads overlapped with current tile's WMMAs.
// ---------------------------------------------------------------------------
__global__ __launch_bounds__(256)
void gemm_edge(const float* __restrict__ A0, int K0,
               const float* __restrict__ A1, int K1,
               const unsigned short* __restrict__ Bt,
               float* __restrict__ C, int M, int K, int Nout) {
  __shared__ __align__(16) unsigned short lA[2][128 * 40];
  __shared__ __align__(16) unsigned short lB[2][64 * 40];
  const int tid = threadIdx.x, lane = tid & 31, wave = tid >> 5;
  const int mBase = blockIdx.x * 128, nBase = blockIdx.y * 64;
  v8f acc[4] = {};

  const int arow = tid >> 1, aseg = (tid & 1) * 16;
  const int bcol = tid >> 2, bks  = (tid & 3) * 8;
  int gr = mBase + arow; if (gr >= M) gr = M - 1;        // clamp: no OOB

  const float* rowA0 = A0 + (size_t)gr * K0;
  const float* rowA1 = A1 + (size_t)gr * K1;
  const unsigned short* gB = Bt + (size_t)(nBase + bcol) * K + bks;

  const unsigned dB[2] = { lds_off_of(&lB[0][bcol * 40 + bks]),
                           lds_off_of(&lB[1][bcol * 40 + bks]) };
  unsigned short* dAp[2] = { &lA[0][arow * 40 + aseg], &lA[1][arow * 40 + aseg] };

  const int nk = K / 32;
  async_b128(dB[0], gB);
  if (nk > 1) async_b128(dB[1], gB + 32);

  // first A tile into registers
  v4f f[4];
  {
    const int g0 = aseg;                                 // kc = 0
    const v4f* s4 = (const v4f*)((g0 < K0) ? (rowA0 + g0) : (rowA1 + g0 - K0));
    f[0] = s4[0]; f[1] = s4[1]; f[2] = s4[2]; f[3] = s4[3];
  }

  for (int i = 0; i < nk; ++i) {
    const int buf = i & 1;
    // convert current A tile
    uint4 u0, u1;
    u0.x = pack_bf16(f[0].x, f[0].y); u0.y = pack_bf16(f[0].z, f[0].w);
    u0.z = pack_bf16(f[1].x, f[1].y); u0.w = pack_bf16(f[1].z, f[1].w);
    u1.x = pack_bf16(f[2].x, f[2].y); u1.y = pack_bf16(f[2].z, f[2].w);
    u1.z = pack_bf16(f[3].x, f[3].y); u1.w = pack_bf16(f[3].z, f[3].w);
    // prefetch next A tile into registers (overlaps compute below)
    if (i + 1 < nk) {
      const int g0 = (i + 1) * 32 + aseg;                // K0 % 16 == 0
      const v4f* s4 = (const v4f*)((g0 < K0) ? (rowA0 + g0) : (rowA1 + g0 - K0));
      f[0] = s4[0]; f[1] = s4[1]; f[2] = s4[2]; f[3] = s4[3];
    }
    unsigned short* d = dAp[buf];
    *(uint4*)d       = u0;
    *(uint4*)(d + 8) = u1;

    if (i + 1 < nk) wait_async1(); else wait_async0();   // B tile i resident
    __syncthreads();
    tile_wmma(lA[buf], lB[buf], lane, wave, acc);
    __syncthreads();
    if (i + 2 < nk) async_b128(dB[buf], gB + (i + 2) * 32);
  }
  epilogue(acc, nullptr, C, mBase, nBase, lane, wave, M, Nout);
}

// ---------------------------------------------------------------------------
__global__ void init_stats(float* amax, float* denom, int n) {
  int i = blockIdx.x * blockDim.x + threadIdx.x;
  if (i < n) { amax[i] = __int_as_float(0xFF800000); denom[i] = 0.0f; }
}

// one wave32 per edge; lanes 0-15 head0, lanes 16-31 head1
__global__ __launch_bounds__(256)
void alpha_kernel(const float* __restrict__ q, const float* __restrict__ k,
                  const float* __restrict__ ebuf,
                  const int* __restrict__ src, const int* __restrict__ dst,
                  float* __restrict__ alpha, float* __restrict__ amax, int E) {
  const int wid  = (blockIdx.x * blockDim.x + threadIdx.x) >> 5;
  const int lane = threadIdx.x & 31;
  if (wid >= E) return;
  const int s = src[wid], d = dst[wid];
  const v4f* qp = (const v4f*)(q    + (size_t)d   * 512);
  const v4f* kp = (const v4f*)(k    + (size_t)s   * 512);
  const v4f* ep = (const v4f*)(ebuf + (size_t)wid * 512);
  const int b4 = lane * 4;
  float accd = 0.f;
#pragma unroll
  for (int i = 0; i < 4; ++i) {
    v4f qv = qp[b4 + i];
    v4f kv = kp[b4 + i];
    v4f ev = __builtin_nontemporal_load(&ep[b4 + i]);
    accd += qv.x * (kv.x + ev.x) + qv.y * (kv.y + ev.y) +
            qv.z * (kv.z + ev.z) + qv.w * (kv.w + ev.w);
  }
#pragma unroll
  for (int off = 8; off >= 1; off >>= 1) accd += __shfl_xor(accd, off, 32);
  if ((lane & 15) == 0) {
    const int h = lane >> 4;
    const float a = accd * 0.0625f;              // 1/sqrt(256)
    alpha[(size_t)wid * 2 + h] = a;
    atomicMaxF(&amax[(size_t)d * 2 + h], a);
  }
}

__global__ void exp_kernel(const int* __restrict__ dst, float* __restrict__ alpha,
                           const float* __restrict__ amax,
                           float* __restrict__ denom, int E2) {
  int i = blockIdx.x * blockDim.x + threadIdx.x;
  if (i >= E2) return;
  const int eid = i >> 1, h = i & 1;
  const int d = dst[eid];
  const float ex = __expf(alpha[i] - amax[(size_t)d * 2 + h]);
  alpha[i] = ex;
  atomicAddF(&denom[(size_t)d * 2 + h], ex);
}

// one wave32 per edge; out[dst] += alpha * (v[src] + e)
__global__ __launch_bounds__(256)
void scatter_kernel(const float* __restrict__ v, const float* __restrict__ ebuf,
                    const float* __restrict__ alpha, const float* __restrict__ denom,
                    const int* __restrict__ src, const int* __restrict__ dst,
                    float* __restrict__ out, int E) {
  const int wid  = (blockIdx.x * blockDim.x + threadIdx.x) >> 5;
  const int lane = threadIdx.x & 31;
  if (wid >= E) return;
  const int s = src[wid], d = dst[wid];
  const int h = lane >> 4;
  const float w = alpha[(size_t)wid * 2 + h] /
                  (denom[(size_t)d * 2 + h] + 1e-16f);
  const v4f* vp = (const v4f*)(v    + (size_t)s   * 512);
  const v4f* ep = (const v4f*)(ebuf + (size_t)wid * 512);
  float* op = out + (size_t)d * 512 + lane * 16;
  const int b4 = lane * 4;
#pragma unroll
  for (int i = 0; i < 4; ++i) {
    v4f vv = vp[b4 + i];
    v4f ee = __builtin_nontemporal_load(&ep[b4 + i]);
    atomicAddF(op + i * 4 + 0, w * (vv.x + ee.x));
    atomicAddF(op + i * 4 + 1, w * (vv.y + ee.y));
    atomicAddF(op + i * 4 + 2, w * (vv.z + ee.z));
    atomicAddF(op + i * 4 + 3, w * (vv.w + ee.w));
  }
}

// ---------------------------------------------------------------------------
extern "C" void kernel_launch(void* const* d_in, const int* in_sizes, int n_in,
                              void* d_out, int out_size, void* d_ws, size_t ws_size,
                              hipStream_t stream) {
  const float* x    = (const float*)d_in[0];
  // d_in[1] = last_update (unused by reference)
  const int*   ei   = (const int*)d_in[2];
  const float* t    = (const float*)d_in[3];
  const float* msg  = (const float*)d_in[4];
  const float* Wq   = (const float*)d_in[5];
  const float* bq   = (const float*)d_in[6];
  const float* Wk   = (const float*)d_in[7];
  const float* bk   = (const float*)d_in[8];
  const float* Wv   = (const float*)d_in[9];
  const float* bv   = (const float*)d_in[10];
  const float* We   = (const float*)d_in[11];
  const float* Wsk  = (const float*)d_in[12];
  const float* bsk  = (const float*)d_in[13];

  const int N = in_sizes[0] / 512;
  const int E = in_sizes[2] / 2;
  const int* srcIdx = ei;
  const int* dstIdx = ei + E;

  char* ws = (char*)d_ws;
  float* q     = (float*)ws; ws += (size_t)N * 512 * sizeof(float);
  float* k     = (float*)ws; ws += (size_t)N * 512 * sizeof(float);
  float* v     = (float*)ws; ws += (size_t)N * 512 * sizeof(float);
  float* ebuf  = (float*)ws; ws += (size_t)E * 512 * sizeof(float);
  float* alpha = (float*)ws; ws += (size_t)E * 2   * sizeof(float);
  float* amax  = (float*)ws; ws += (size_t)N * 2   * sizeof(float);
  float* denom = (float*)ws; ws += (size_t)N * 2   * sizeof(float);
  unsigned short* xb  = (unsigned short*)ws; ws += (size_t)N * 512 * 2;
  unsigned short* wtq = (unsigned short*)ws; ws += (size_t)512 * 512 * 2;
  unsigned short* wtk = (unsigned short*)ws; ws += (size_t)512 * 512 * 2;
  unsigned short* wtv = (unsigned short*)ws; ws += (size_t)512 * 512 * 2;
  unsigned short* wts = (unsigned short*)ws; ws += (size_t)512 * 512 * 2;
  unsigned short* wte = (unsigned short*)ws; ws += (size_t)512 * 384 * 2;
  float* out   = (float*)d_out;

  const dim3 blk(256);

  // ---- one-time pre-conversion: x -> bf16, weights -> bf16 transposed ----
  {
    size_t nx = (size_t)N * 512;
    cvt_f32_to_bf16<<<(unsigned)((nx / 8 + 255) / 256), blk, 0, stream>>>(x, xb, nx);
    int nw = 512 * 512;
    transpose_cvt<<<(nw + 255) / 256, blk, 0, stream>>>(Wq,  wtq, 512, 512);
    transpose_cvt<<<(nw + 255) / 256, blk, 0, stream>>>(Wk,  wtk, 512, 512);
    transpose_cvt<<<(nw + 255) / 256, blk, 0, stream>>>(Wv,  wtv, 512, 512);
    transpose_cvt<<<(nw + 255) / 256, blk, 0, stream>>>(Wsk, wts, 512, 512);
    int ne = 384 * 512;
    transpose_cvt<<<(ne + 255) / 256, blk, 0, stream>>>(We,  wte, 384, 512);
  }

  // ---- node GEMMs: q, k, v; skip straight into d_out ---------------------
  dim3 gN((N + 127) / 128, 512 / 64);
  gemm_node<<<gN, blk, 0, stream>>>(xb, wtq, bq,  q,   N, 512, 512);
  gemm_node<<<gN, blk, 0, stream>>>(xb, wtk, bk,  k,   N, 512, 512);
  gemm_node<<<gN, blk, 0, stream>>>(xb, wtv, bv,  v,   N, 512, 512);
  gemm_node<<<gN, blk, 0, stream>>>(xb, wts, bsk, out, N, 512, 512);

  // ---- edge GEMM: e = concat(t, msg) @ We --------------------------------
  dim3 gE((E + 127) / 128, 512 / 64);
  gemm_edge<<<gE, blk, 0, stream>>>(t, 128, msg, 256, wte, ebuf, E, 384, 512);

  init_stats<<<(2 * N + 255) / 256, blk, 0, stream>>>(amax, denom, 2 * N);

  const int edgeBlocks = (E + 7) / 8;   // 8 wave32 edges per 256-thread block
  alpha_kernel<<<edgeBlocks, blk, 0, stream>>>(q, k, ebuf, srcIdx, dstIdx, alpha, amax, E);
  exp_kernel<<<(2 * E + 255) / 256, blk, 0, stream>>>(dstIdx, alpha, amax, denom, 2 * E);
  scatter_kernel<<<edgeBlocks, blk, 0, stream>>>(v, ebuf, alpha, denom, srcIdx, dstIdx, out, E);
}
